// LSTMx_77283641524385
// MI455X (gfx1250) — matrix operand order repaired and checked by reading, not compiled, not run on hardware
//
#include <hip/hip_runtime.h>

// ---------------------------------------------------------------------------
// 2-layer LSTM for MI455X (gfx1250, wave32, WMMA).
// - Recurrent + input GEMM fused on v_wmma_f32_16x16x32_bf16 (f32 accum).
// - Cell state c lives in accumulator-layout VGPRs across all 512 timesteps.
// - Persistent kernel, one atomic grid barrier per timestep.
// - h(t) staged once per WG into LDS via global_load_async_to_lds_b128
//   (ASYNCcnt / s_wait_asynccnt), WMMA A-fragments then come from LDS
//   (ds_load_b128, bank-conflict-free via +8 bf16 row pad).
// - Weights (27MB bf16 total) stream from the 192MB L2 every step.
// ---------------------------------------------------------------------------

static constexpr int SEQ = 512;   // L
static constexpr int NB  = 64;    // batch
static constexpr int DIN = 256;   // layer-0 input size
static constexpr int HD  = 1024;  // hidden
static constexpr int NWG = 32;    // persistent workgroups (grid barrier)
static constexpr int BLK = 256;   // 8 waves of 32
static constexpr int HSTR = HD + 8;  // LDS row stride (elems): 2064B -> conflict-free b128

typedef __attribute__((ext_vector_type(16))) __bf16 v16bf;
typedef __attribute__((ext_vector_type(8)))  float  v8f;

__device__ inline __bf16 f2bf(float f) {
  unsigned u = __builtin_bit_cast(unsigned, f);
  u += 0x7FFFu + ((u >> 16) & 1u);           // round-to-nearest-even
  unsigned short h = (unsigned short)(u >> 16);
  return __builtin_bit_cast(__bf16, h);
}

__device__ inline float sigm(float x) { return 1.0f / (1.0f + __expf(-x)); }

// A-fragment (16x32 bf16, M=lane&15, K-halves per ISA layout):
// elems 0..7  -> K = kb + 8*half + 0..7
// elems 8..15 -> K = kb + 16 + 8*half + 0..7
__device__ inline v16bf load_a32(const __bf16* p /* = row + kb + 8*half */) {
  union { v16bf v; uint4 q[2]; } u;
  u.q[0] = *(const uint4*)(p);
  u.q[1] = *(const uint4*)(p + 16);
  return u.v;
}

// B-fragment (32x16 bf16, N=lane&15), same K order as A:
// elems 0..7 -> K = kb + 8*half + 0..7, elems 8..15 -> K = kb + 16 + 8*half + 0..7.
__device__ inline v16bf load_b32(const __bf16* p /* = wrow + kb + 8*half */) {
  union { v16bf v; uint4 q[2]; } u;
  u.q[0] = *(const uint4*)(p);
  u.q[1] = *(const uint4*)(p + 16);
  return u.v;
}

__global__ void init_kernel(int* s) { s[0] = 0; s[1] = 0; }

__global__ void cvt_kernel(const float* __restrict__ s, __bf16* __restrict__ d, int n) {
  for (int i = blockIdx.x * blockDim.x + threadIdx.x; i < n; i += gridDim.x * blockDim.x)
    d[i] = f2bf(s[i]);
}

__global__ void bias_kernel(const float* __restrict__ a, const float* __restrict__ b,
                            float* __restrict__ o, int n) {
  for (int i = blockIdx.x * blockDim.x + threadIdx.x; i < n; i += gridDim.x * blockDim.x)
    o[i] = a[i] + b[i];
}

__global__ __launch_bounds__(BLK)
void lstm_layer_kernel(const __bf16* __restrict__ whh,   // (4H, H) bf16
                       const __bf16* __restrict__ wih,   // (4H, din) bf16
                       int din,
                       __bf16* __restrict__ hsbf,        // (L+1, N, H); slot0 = h0
                       const __bf16* __restrict__ xseq,  // (L, N, din)
                       const float*  __restrict__ bsum,  // (4H) = b_ih + b_hh
                       const float*  __restrict__ c0,    // (N, H)
                       float* __restrict__ outh,         // null (layer0) or (L,N,H)
                       float* __restrict__ allc,         // (L, 2, N, H) region
                       float* __restrict__ hn,           // (N, H)
                       float* __restrict__ cn,           // (N, H)
                       int layer,
                       int* __restrict__ sync) {
  __shared__ __bf16 hbuf[NB * HSTR];   // staged h(t), ~129KB of the 320KB WGP LDS

  const int tid  = threadIdx.x;
  const int wave = tid >> 5;
  const int lane = tid & 31;
  const int half = lane >> 4;       // lane-half
  const int nl   = lane & 15;
  const int jt   = wave & 1;        // j-tile within WG
  const int bt   = wave >> 1;       // batch-tile
  const int j0   = blockIdx.x * 32 + jt * 16;
  const int jcol = j0 + nl;         // this lane's output column (C/D: N = lane&15)

  // Per-gate B row pointers + bias scalars (constant over time).
  const __bf16* bh[4];
  const __bf16* bx[4];
  float bias[4];
#pragma unroll
  for (int g = 0; g < 4; ++g) {
    size_t row = (size_t)(g * HD + j0 + nl);
    bh[g]   = whh + row * (size_t)HD  + 8 * half;
    bx[g]   = wih + row * (size_t)din + 8 * half;
    bias[g] = bsum[g * HD + jcol];
  }

  // A row offsets (M = lane&15 within the batch tile).
  const int    am    = bt * 16 + nl;
  const __bf16* ahl  = hbuf + am * HSTR + 8 * half;   // LDS A row (h part)
  const size_t aoffX = (size_t)am * din + 8 * half;   // global A row (x part)

  // Cell state lives in accumulator layout: elem r -> batch n = bt*16 + r + 8*half.
  v8f c;
#pragma unroll
  for (int r = 0; r < 8; ++r) {
    int n = bt * 16 + r + 8 * half;
    c[r] = c0[(size_t)n * HD + jcol];
  }

  const v8f vz = {0.f, 0.f, 0.f, 0.f, 0.f, 0.f, 0.f, 0.f};

  for (int t = 0; t < SEQ; ++t) {
    // ---- Stage h(t) (64x1024 bf16 = 128KB) into LDS with async copies. ----
    // Generic pointers into __shared__ carry the LDS byte offset in their low
    // 32 bits (flat->LDS aperture truncation), which is exactly what the
    // async instruction's VDST operand wants.
    {
      const __bf16* hsrc = hsbf + (size_t)t * NB * HD;
      for (int ci = tid; ci < NB * HD / 8; ci += BLK) {   // 8192 16B chunks
        int row  = ci >> 7;                                // 128 chunks per row
        int cole = (ci & 127) << 3;                        // element offset
        unsigned loff = (unsigned)(unsigned long long)(const void*)&hbuf[row * HSTR + cole];
        const __bf16* g = hsrc + row * HD + cole;
        asm volatile("global_load_async_to_lds_b128 %0, %1, off"
                     :: "v"(loff), "v"(g) : "memory");
      }
      asm volatile("s_wait_asynccnt 0x0" ::: "memory");
      __syncthreads();
    }

    const __bf16* ax = xseq + (size_t)t * NB * din + aoffX;

    v8f acc[4] = {vz, vz, vz, vz};   // i, f, g, o

    // Recurrent part: K = H over W_hh; A from LDS, B streamed from L2.
    for (int kb = 0; kb < HD; kb += 32) {
      v16bf a = load_a32(ahl + kb);
#pragma unroll
      for (int g = 0; g < 4; ++g) {
        v16bf b = load_b32(bh[g] + kb);
        acc[g] = __builtin_amdgcn_wmma_f32_16x16x32_bf16(
            false, a, false, b, (short)0, acc[g], false, false);
      }
    }
    // Input part: K = din over W_ih (folded into the same accumulators).
    for (int kb = 0; kb < din; kb += 32) {
      v16bf a = load_a32(ax + kb);
#pragma unroll
      for (int g = 0; g < 4; ++g) {
        v16bf b = load_b32(bx[g] + kb);
        acc[g] = __builtin_amdgcn_wmma_f32_16x16x32_bf16(
            false, a, false, b, (short)0, acc[g], false, false);
      }
    }

    __bf16* hdst = hsbf + (size_t)(t + 1) * NB * HD;
#pragma unroll
    for (int r = 0; r < 8; ++r) {
      int   n  = bt * 16 + r + 8 * half;
      float iv = sigm(acc[0][r] + bias[0]);
      float fv = sigm(acc[1][r] + bias[1]);
      float gv = tanhf(acc[2][r] + bias[2]);
      float ov = sigm(acc[3][r] + bias[3]);
      float cv = fv * c[r] + iv * gv;
      c[r] = cv;
      float hv = ov * tanhf(cv);

      allc[((size_t)(t * 2 + layer) * NB + n) * HD + jcol] = cv;
      hdst[(size_t)n * HD + jcol] = f2bf(hv);
      if (outh) outh[((size_t)t * NB + n) * HD + jcol] = hv;
      if (t == SEQ - 1) {
        hn[(size_t)n * HD + jcol] = hv;
        cn[(size_t)n * HD + jcol] = cv;
      }
    }

    // Grid-wide barrier: h(t) must be globally visible before step t+1.
    __syncthreads();
    __threadfence();
    if (tid == 0) {
      int gen = atomicAdd(sync + 1, 0);
      if (atomicAdd(sync, 1) == (int)gridDim.x - 1) {
        atomicExch(sync, 0);
        __threadfence();
        atomicAdd(sync + 1, 1);
      } else {
        while (atomicAdd(sync + 1, 0) == gen) __builtin_amdgcn_s_sleep(2);
      }
    }
    __syncthreads();
  }
}

extern "C" void kernel_launch(void* const* d_in, const int* in_sizes, int n_in,
                              void* d_out, int out_size, void* d_ws, size_t ws_size,
                              hipStream_t stream) {
  const float* x    = (const float*)d_in[0];
  const float* h0   = (const float*)d_in[1];
  const float* c0   = (const float*)d_in[2];
  const float* wih0 = (const float*)d_in[3];
  const float* whh0 = (const float*)d_in[4];
  const float* bih0 = (const float*)d_in[5];
  const float* bhh0 = (const float*)d_in[6];
  const float* wih1 = (const float*)d_in[7];
  const float* whh1 = (const float*)d_in[8];
  const float* bih1 = (const float*)d_in[9];
  const float* bhh1 = (const float*)d_in[10];
  float* out = (float*)d_out;

  char*  ws  = (char*)d_ws;
  int*   syn = (int*)ws;
  size_t off = 256;
  auto alloc_bf = [&](size_t nelem) {
    __bf16* p = (__bf16*)(ws + off);
    off = (off + nelem * 2 + 255) & ~(size_t)255;
    return p;
  };
  __bf16* xbf   = alloc_bf((size_t)SEQ * NB * DIN);
  __bf16* wih0b = alloc_bf((size_t)4 * HD * DIN);
  __bf16* whh0b = alloc_bf((size_t)4 * HD * HD);
  __bf16* wih1b = alloc_bf((size_t)4 * HD * HD);
  __bf16* whh1b = alloc_bf((size_t)4 * HD * HD);
  __bf16* hs0   = alloc_bf((size_t)(SEQ + 1) * NB * HD);
  __bf16* hs1   = alloc_bf((size_t)(SEQ + 1) * NB * HD);
  float* bsum0 = (float*)(ws + off); off += (size_t)4 * HD * 4;
  float* bsum1 = (float*)(ws + off); off += (size_t)4 * HD * 4;

  init_kernel<<<1, 1, 0, stream>>>(syn);
  cvt_kernel<<<1024, 256, 0, stream>>>(x,    xbf,   SEQ * NB * DIN);
  cvt_kernel<<<1024, 256, 0, stream>>>(wih0, wih0b, 4 * HD * DIN);
  cvt_kernel<<<1024, 256, 0, stream>>>(whh0, whh0b, 4 * HD * HD);
  cvt_kernel<<<1024, 256, 0, stream>>>(wih1, wih1b, 4 * HD * HD);
  cvt_kernel<<<1024, 256, 0, stream>>>(whh1, whh1b, 4 * HD * HD);
  cvt_kernel<<<64,   256, 0, stream>>>(h0,           hs0, NB * HD);   // h0 layer0 -> slot 0
  cvt_kernel<<<64,   256, 0, stream>>>(h0 + NB * HD, hs1, NB * HD);   // h0 layer1 -> slot 0
  bias_kernel<<<16, 256, 0, stream>>>(bih0, bhh0, bsum0, 4 * HD);
  bias_kernel<<<16, 256, 0, stream>>>(bih1, bhh1, bsum1, 4 * HD);

  float* allc = out  + (size_t)SEQ * NB * HD;            // (L, 2, N, H)
  float* hn   = allc + (size_t)SEQ * 2 * NB * HD;        // (2, N, H)
  float* cn   = hn   + (size_t)2 * NB * HD;              // (2, N, H)

  // Layer 0: input = x (din=256); h-sequence -> hs0 (bf16, feeds layer 1).
  lstm_layer_kernel<<<NWG, BLK, 0, stream>>>(
      whh0b, wih0b, DIN, hs0, xbf, bsum0, c0, nullptr,
      allc, hn, cn, 0, syn);

  // Layer 1: input = hs0[t] (= slot t+1 of hs0); h -> output region (fp32) + hs1 (bf16).
  lstm_layer_kernel<<<NWG, BLK, 0, stream>>>(
      whh1b, wih1b, HD, hs1, hs0 + (size_t)NB * HD, bsum1, c0 + (size_t)NB * HD, out,
      allc, hn + (size_t)NB * HD, cn + (size_t)NB * HD, 1, syn);
}
